// CANModel_59717225283740
// MI455X (gfx1250) — matrix operand order repaired and checked by reading, not compiled, not run on hardware
//
#include <hip/hip_runtime.h>

#define N_NODES 131072
#define HEADS   4
#define DHEAD   32
#define CDIM    128
#define ENCDIM  32
#define NNZ_UP  (4 * N_NODES)
#define EPS_SKIP 1.000001f

typedef __attribute__((ext_vector_type(16))) _Float16 v16h;
typedef __attribute__((ext_vector_type(8)))  _Float16 v8h;
typedef __attribute__((ext_vector_type(4)))  _Float16 v4h;
typedef __attribute__((ext_vector_type(8)))  float    v8f;
typedef __attribute__((ext_vector_type(4)))  unsigned v4ui;
typedef __attribute__((ext_vector_type(8)))  int      v8i;
typedef __attribute__((ext_vector_type(4)))  int      v4i;

// Toolchain discrimination: the gfx1250 TDM header ships only with the
// amdgpu-toolchain (6-arg tensor_load_to_lds builtin); ROCm 7.2 is 5-arg.
#if __has_include(<hip/amd_detail/amd_gfx1250_TDM.h>)
#define TDM_6ARG 1
#else
#define TDM_6ARG 0
#endif

// ---------------------------------------------------------------------------
// TDM: DMA a [rows x width_elems] 2-D tile of 2-byte elements from global
// memory into LDS. D# bit layout per CDNA5 ISA ch.8 (group0/group1).
// Issue from a single wave; EXEC is ignored by tensor ops.
// ---------------------------------------------------------------------------
__device__ __forceinline__ void tdm_load_tile_to_lds(void* lds_ptr, const void* gptr,
                                                     int width_elems, int rows,
                                                     int tensor_rows) {
  unsigned long long ga = (unsigned long long)(uintptr_t)gptr;
  unsigned ldsa = (unsigned)(uintptr_t)lds_ptr;   // low 32 bits of flat LDS addr
  v4ui g0;
  g0[0] = 1u;                                   // count=1, user descriptor
  g0[1] = ldsa;                                 // lds_addr
  g0[2] = (unsigned)ga;                         // global_addr[31:0]
  g0[3] = (unsigned)(ga >> 32) | (2u << 30);    // global_addr[56:32] | type=2
  v8i g1;
  g1[0] = 1 << 16;                                                   // data_size=2B
  g1[1] = (width_elems & 0xffff) << 16;                              // tensor_dim0 lo16
  g1[2] = ((width_elems >> 16) & 0xffff) | ((tensor_rows & 0xffff) << 16);
  g1[3] = ((tensor_rows >> 16) & 0xffff) | ((width_elems & 0xffff) << 16); // tile_dim0
  g1[4] = rows & 0xffff;                                             // tile_dim1
  g1[5] = width_elems;                                               // dim0_stride lo32
  g1[6] = 0;
  g1[7] = 0;
  v4i z4 = {0, 0, 0, 0};
#if TDM_6ARG
  v8i z8 = {0, 0, 0, 0, 0, 0, 0, 0};
  __builtin_amdgcn_tensor_load_to_lds(g0, g1, z4, z4, z8, 0);
#else
  __builtin_amdgcn_tensor_load_to_lds(g0, g1, z4, z4, 0);
#endif
}

// ---------------------------------------------------------------------------
// Utility kernels
// ---------------------------------------------------------------------------
__global__ void fill_kernel(float* __restrict__ p, float v, int n) {
  int i = blockIdx.x * blockDim.x + threadIdx.x;
  if (i < n) p[i] = v;
}

__global__ void relu_kernel(const float* __restrict__ in, float* __restrict__ out, int n) {
  int i = blockIdx.x * blockDim.x + threadIdx.x;
  if (i < n) {
    float v = in[i];
    out[i] = v > 0.f ? v : 0.f;
  }
}

// f32 -> f16, vectorized x4 (n4 = count/4)
__global__ void cvt_f16_kernel(const float* __restrict__ in, _Float16* __restrict__ out,
                               int n4) {
  int i = blockIdx.x * blockDim.x + threadIdx.x;
  if (i < n4) {
    float4 v = ((const float4*)in)[i];
    v4h h;
    h[0] = (_Float16)v.x; h[1] = (_Float16)v.y;
    h[2] = (_Float16)v.z; h[3] = (_Float16)v.w;
    ((v4h*)out)[i] = h;
  }
}

// W[K,M] f32 -> Wt[M,K] f16 (transpose + convert)
__global__ void cvt_f16_t_kernel(const float* __restrict__ W, _Float16* __restrict__ Wt,
                                 int K, int M) {
  int i = blockIdx.x * blockDim.x + threadIdx.x;
  if (i < K * M) {
    int k = i / M, m = i - k * M;
    Wt[(size_t)m * K + k] = (_Float16)W[i];
  }
}

// Column sums of an [nrows, 128] matrix. blockDim.x == 128.
__global__ void colsum_kernel(const float* __restrict__ x, float* __restrict__ cs, int nrows) {
  int c = threadIdx.x;
  float s = 0.f;
  for (int r = blockIdx.x; r < nrows; r += gridDim.x)
    s += x[(size_t)r * CDIM + c];
  atomicAdd(&cs[c], s);
}

// ---------------------------------------------------------------------------
// WMMA GEMM:  Y[N_NODES, 128] = scale * (Ah[N_NODES, K] @ Wt^T) + bias
// Ah is f16 row-major, Wt is f16 transposed [128, K]. One block = one 16-row
// strip; 8 waves = the 8 column tiles; A strip staged to LDS via TDM and
// reused by all waves. K in {32, 128}, fully unrolled WMMA chain.
// ---------------------------------------------------------------------------
template <int K>
__global__ __launch_bounds__(256) void gemm_wmma_kernel(
    const _Float16* __restrict__ Ah, const _Float16* __restrict__ Wt,
    const float* __restrict__ bias, float* __restrict__ Y, float scale) {
  __shared__ __align__(16) _Float16 Als[16 * K];
  const int blockRow = blockIdx.x;

  if ((threadIdx.x >> 5) == 0) {   // wave 0 issues the tensor DMA
    tdm_load_tile_to_lds(Als, Ah + (size_t)blockRow * 16 * K, K, 16, N_NODES);
    __builtin_amdgcn_s_wait_tensorcnt(0);
  }
  __syncthreads();

  const int lane    = threadIdx.x & 31;
  const int tileCol = threadIdx.x >> 5;   // 0..7 -> output cols 16*tileCol..
  const bool hi     = lane >= 16;
  const int  l16    = lane & 15;
  const int  bcol   = tileCol * 16 + l16;

  v8f acc = {};
#pragma unroll
  for (int kc = 0; kc < K; kc += 32) {
    // A 16x32 f16 fragment (ISA layout): lanes 0-15 hold K=kc+[0..7],[16..23];
    // lanes 16-31 hold K=kc+[8..15],[24..31] of row l16. From LDS (ds_load_b128).
    const _Float16* ap = &Als[l16 * K + kc + (hi ? 8 : 0)];
    v8h a0 = *(const v8h*)ap;
    v8h a1 = *(const v8h*)(ap + 16);
    // B 32x16 fragment: lane holds column bcol, K=kc+[0..7],[16..23] (lanes 0-15) or
    // kc+[8..15],[24..31] (lanes 16-31), matching A.
    const _Float16* bp = Wt + (size_t)bcol * K + kc + (hi ? 8 : 0);
    v8h b0 = *(const v8h*)bp;
    v8h b1 = *(const v8h*)(bp + 16);
    v16h a, b;
#pragma unroll
    for (int i = 0; i < 8; ++i) {
      a[i] = a0[i]; a[i + 8] = a1[i];
      b[i] = b0[i]; b[i + 8] = b1[i];
    }
    acc = __builtin_amdgcn_wmma_f32_16x16x32_f16(
        false, a, false, b, (short)0, acc, false, false);
  }

  // C/D layout: VGPR r -> row r (lanes 0-15) / r+8 (lanes 16-31), col lane%16.
  float bv = bias ? bias[bcol] : 0.0f;
#pragma unroll
  for (int r = 0; r < 8; ++r) {
    int orow = blockRow * 16 + r + (hi ? 8 : 0);
    Y[(size_t)orow * CDIM + bcol] = acc[r] * scale + bv;
  }
}

// ---------------------------------------------------------------------------
// Lift layer
// ---------------------------------------------------------------------------
__global__ void lift_kernel(const float* __restrict__ x0t,
                            const float* __restrict__ x1lin,
                            const float* __restrict__ att_lift,
                            const int* __restrict__ src, const int* __restrict__ dst,
                            float* __restrict__ xout) {
  int idx = blockIdx.x * blockDim.x + threadIdx.x;
  if (idx >= N_NODES * HEADS) return;
  int n = idx >> 2, h = idx & 3;
  int si = src[n], di = dst[n];
  const float* xs = x0t + (size_t)si * CDIM + h * DHEAD;
  const float* xd = x0t + (size_t)di * CDIM + h * DHEAD;
  const float* al = att_lift + h * (2 * DHEAD);
  float s = 0.f;
#pragma unroll 8
  for (int d = 0; d < DHEAD; ++d) s += xs[d] * al[d] + xd[d] * al[DHEAD + d];
  s = s > 0.f ? s : 0.f;
  const float* x1 = x1lin + (size_t)n * CDIM + h * DHEAD;
  float* xo = xout + (size_t)n * CDIM + h * DHEAD;
#pragma unroll 8
  for (int d = 0; d < DHEAD; ++d) xo[d] = s * x1[d];
}

// ---------------------------------------------------------------------------
// Attention machinery
// ---------------------------------------------------------------------------
__global__ void attn_proj_kernel(const float* __restrict__ xh,
                                 const float* __restrict__ asrc,
                                 const float* __restrict__ adst,
                                 float* __restrict__ as_out, float* __restrict__ ad_out,
                                 int n) {
  int idx = blockIdx.x * blockDim.x + threadIdx.x;
  if (idx >= n * HEADS) return;
  int node = idx >> 2, h = idx & 3;
  const float* xr = xh + (size_t)node * CDIM + h * DHEAD;
  const float* s  = asrc + h * DHEAD;
  const float* d  = adst + h * DHEAD;
  float vs = 0.f, vd = 0.f;
#pragma unroll 8
  for (int i = 0; i < DHEAD; ++i) {
    float x = xr[i];
    vs += x * s[i];
    vd += x * d[i];
  }
  as_out[idx] = vs;
  ad_out[idx] = vd;
}

__device__ __forceinline__ float leaky02(float e) { return e > 0.f ? e : 0.2f * e; }

__device__ __forceinline__ void atomicMaxF(float* addr, float val) {
  if (val >= 0.f) atomicMax((int*)addr, __float_as_int(val));
  else            atomicMin((unsigned int*)addr, __float_as_uint(val));
}

__global__ void edge_max_kernel(const float* __restrict__ a_s, const float* __restrict__ a_d,
                                const int* __restrict__ src, const int* __restrict__ dst,
                                float* __restrict__ m, int nnz) {
  int idx = blockIdx.x * blockDim.x + threadIdx.x;
  if (idx >= nnz * HEADS) return;
  int i = idx >> 2, h = idx & 3;
  int di = dst[i];
  float e = leaky02(a_s[src[i] * HEADS + h] + a_d[di * HEADS + h]);
  atomicMaxF(&m[di * HEADS + h], e);
}

__global__ void edge_expsum_kernel(const float* __restrict__ a_s, const float* __restrict__ a_d,
                                   const int* __restrict__ src, const int* __restrict__ dst,
                                   const float* __restrict__ m, float* __restrict__ ssum,
                                   int nnz) {
  int idx = blockIdx.x * blockDim.x + threadIdx.x;
  if (idx >= nnz * HEADS) return;
  int i = idx >> 2, h = idx & 3;
  int di = dst[i];
  float e = leaky02(a_s[src[i] * HEADS + h] + a_d[di * HEADS + h]);
  atomicAdd(&ssum[di * HEADS + h], expf(e - m[di * HEADS + h]));
}

__global__ void edge_scatter_kernel(const float* __restrict__ xh,
                                    const float* __restrict__ a_s, const float* __restrict__ a_d,
                                    const int* __restrict__ src, const int* __restrict__ dst,
                                    const float* __restrict__ m, const float* __restrict__ ssum,
                                    float* __restrict__ acc, int nnz) {
  int idx = blockIdx.x * blockDim.x + threadIdx.x;
  if (idx >= nnz * HEADS) return;
  int i = idx >> 2, h = idx & 3;
  int si = src[i], di = dst[i];
  float e = leaky02(a_s[si * HEADS + h] + a_d[di * HEADS + h]);
  float alpha = expf(e - m[di * HEADS + h]) / (ssum[di * HEADS + h] + 1e-16f);
  const float* xs = xh + (size_t)si * CDIM + h * DHEAD;
  float* ac = acc + (size_t)di * CDIM + h * DHEAD;
#pragma unroll 8
  for (int d = 0; d < DHEAD; ++d) atomicAdd(&ac[d], xs[d] * alpha);
}

// out = b0 + b1 + b2 + (colsum0 . w0 + colsum1 . w1) / N
__global__ void final_kernel(const float* __restrict__ colsum0, const float* __restrict__ colsum1,
                             const float* __restrict__ w0, const float* __restrict__ b0,
                             const float* __restrict__ w1, const float* __restrict__ b1,
                             const float* __restrict__ b2, float* __restrict__ out) {
  if (threadIdx.x == 0 && blockIdx.x == 0) {
    float acc = 0.f;
    for (int c = 0; c < CDIM; ++c) acc += colsum0[c] * w0[c] + colsum1[c] * w1[c];
    out[0] = b0[0] + b1[0] + b2[0] + acc / (float)N_NODES;
  }
}

// ---------------------------------------------------------------------------
// Orchestration
// ---------------------------------------------------------------------------
extern "C" void kernel_launch(void* const* d_in, const int* in_sizes, int n_in,
                              void* d_out, int out_size, void* d_ws, size_t ws_size,
                              hipStream_t stream) {
  (void)in_sizes; (void)n_in; (void)out_size; (void)ws_size;

  const float* x_0      = (const float*)d_in[0];
  const float* x_1      = (const float*)d_in[1];
  const float* W0in     = (const float*)d_in[2];
  const float* b0in     = (const float*)d_in[3];
  const float* W1in     = (const float*)d_in[4];
  const float* b1in     = (const float*)d_in[5];
  const float* Wpre     = (const float*)d_in[6];
  const float* bpre     = (const float*)d_in[7];
  const float* att_lift = (const float*)d_in[8];
  const float* Wl       = (const float*)d_in[9];
  const float* al_src   = (const float*)d_in[10];
  const float* al_dst   = (const float*)d_in[11];
  const float* Wu       = (const float*)d_in[12];
  const float* au_src   = (const float*)d_in[13];
  const float* au_dst   = (const float*)d_in[14];
  const float* Wskip    = (const float*)d_in[15];
  const float* w0       = (const float*)d_in[16];
  const float* b0       = (const float*)d_in[17];
  const float* w1       = (const float*)d_in[18];
  const float* b1       = (const float*)d_in[19];
  const float* b2       = (const float*)d_in[21];   // w2 (d_in[20]) multiplies zeros
  const int*   lower_src = (const int*)d_in[22];
  const int*   lower_dst = (const int*)d_in[23];
  const int*   upper_src = (const int*)d_in[24];
  const int*   upper_dst = (const int*)d_in[25];

  float* ws = (float*)d_ws;
  const size_t SZ = (size_t)N_NODES * CDIM;
  float* buf0 = ws;                 // x0 -> x (current features)
  float* buf1 = ws + SZ;            // x0t -> xh_lower
  float* buf2 = ws + 2 * SZ;        // x1lin -> xh_upper
  float* buf3 = ws + 3 * SZ;        // skip accumulator
  float* a_s     = ws + 4 * SZ;
  float* a_d     = a_s  + (size_t)N_NODES * HEADS;
  float* mseg    = a_d  + (size_t)N_NODES * HEADS;
  float* sseg    = mseg + (size_t)N_NODES * HEADS;
  float* colsum0 = sseg + (size_t)N_NODES * HEADS;
  float* colsum1 = colsum0 + CDIM;
  // f16 region (16B-aligned by construction of the float offsets above)
  _Float16* Ah = (_Float16*)(colsum1 + CDIM);        // current A operand, f16
  _Float16* wt = Ah + SZ;                            // transposed f16 weights
  _Float16* W0in_t = wt;  wt += (size_t)ENCDIM * CDIM;
  _Float16* W1in_t = wt;  wt += (size_t)ENCDIM * CDIM;
  _Float16* Wpre_t = wt;  wt += (size_t)CDIM * CDIM;
  _Float16* Wl_t   = wt;  wt += (size_t)2 * CDIM * CDIM;
  _Float16* Wu_t   = wt;  wt += (size_t)2 * CDIM * CDIM;
  _Float16* Wsk_t  = wt;  wt += (size_t)2 * CDIM * CDIM;

  const int NH = N_NODES * HEADS;
  const float NEG_INF = -__builtin_inff();

  auto cvtT = [&](const float* W, _Float16* Wt_, int K) {
    int n = K * CDIM;
    cvt_f16_t_kernel<<<(n + 255) / 256, 256, 0, stream>>>(W, Wt_, K, CDIM);
  };
  auto cvtA = [&](const float* X, int cols) {
    int n4 = N_NODES * cols / 4;
    cvt_f16_kernel<<<(n4 + 255) / 256, 256, 0, stream>>>(X, Ah, n4);
  };
  auto gemm32 = [&](const _Float16* Wt_, const float* bias, float* Y) {
    gemm_wmma_kernel<ENCDIM><<<N_NODES / 16, 256, 0, stream>>>(Ah, Wt_, bias, Y, 1.f);
  };
  auto gemm128 = [&](const _Float16* Wt_, const float* bias, float* Y, float scale) {
    gemm_wmma_kernel<CDIM><<<N_NODES / 16, 256, 0, stream>>>(Ah, Wt_, bias, Y, scale);
  };
  auto attention = [&](const float* xh, const float* asrc, const float* adst,
                       const int* esrc, const int* edst, int nnz, float* acc) {
    fill_kernel<<<(NH + 255) / 256, 256, 0, stream>>>(mseg, NEG_INF, NH);
    fill_kernel<<<(NH + 255) / 256, 256, 0, stream>>>(sseg, 0.f, NH);
    attn_proj_kernel<<<(NH + 255) / 256, 256, 0, stream>>>(xh, asrc, adst, a_s, a_d, N_NODES);
    int eh = nnz * HEADS;
    edge_max_kernel<<<(eh + 255) / 256, 256, 0, stream>>>(a_s, a_d, esrc, edst, mseg, nnz);
    edge_expsum_kernel<<<(eh + 255) / 256, 256, 0, stream>>>(a_s, a_d, esrc, edst, mseg, sseg, nnz);
    edge_scatter_kernel<<<(eh + 255) / 256, 256, 0, stream>>>(xh, a_s, a_d, esrc, edst,
                                                              mseg, sseg, acc, nnz);
  };

  // ---- stage all weights as transposed f16 (tiny, once per launch) ----
  cvtT(W0in, W0in_t, ENCDIM);
  cvtT(W1in, W1in_t, ENCDIM);
  cvtT(Wpre, Wpre_t, CDIM);
  for (int l = 0; l < 2; ++l) {
    cvtT(Wl    + (size_t)l * CDIM * CDIM, Wl_t  + (size_t)l * CDIM * CDIM, CDIM);
    cvtT(Wu    + (size_t)l * CDIM * CDIM, Wu_t  + (size_t)l * CDIM * CDIM, CDIM);
    cvtT(Wskip + (size_t)l * CDIM * CDIM, Wsk_t + (size_t)l * CDIM * CDIM, CDIM);
  }

  // ---- encoders ----
  cvtA(x_0, ENCDIM);
  gemm32(W0in_t, b0in, buf0);                                     // x0
  fill_kernel<<<1, 256, 0, stream>>>(colsum0, 0.f, CDIM);
  colsum_kernel<<<1024, 128, 0, stream>>>(buf0, colsum0, N_NODES);
  cvtA(buf0, CDIM);
  gemm128(Wpre_t, bpre, buf1, 1.f);                               // x0t
  cvtA(x_1, ENCDIM);
  gemm32(W1in_t, b1in, buf2);                                     // x1lin

  // ---- lift ----
  lift_kernel<<<(NH + 255) / 256, 256, 0, stream>>>(buf1, buf2, att_lift,
                                                    lower_src, lower_dst, buf0);

  // ---- CAN layers ----
  for (int l = 0; l < 2; ++l) {
    const float* als_l = al_src + (size_t)l * HEADS * DHEAD;
    const float* ald_l = al_dst + (size_t)l * HEADS * DHEAD;
    const float* aus_l = au_src + (size_t)l * HEADS * DHEAD;
    const float* aud_l = au_dst + (size_t)l * HEADS * DHEAD;

    cvtA(buf0, CDIM);  // x in f16, shared by the 3 GEMMs below
    gemm128(Wl_t  + (size_t)l * CDIM * CDIM, nullptr, buf1, 1.f);       // xh_lower
    gemm128(Wu_t  + (size_t)l * CDIM * CDIM, nullptr, buf2, 1.f);       // xh_upper
    gemm128(Wsk_t + (size_t)l * CDIM * CDIM, nullptr, buf3, EPS_SKIP);  // EPS*(x@Wskip)

    attention(buf1, als_l, ald_l, lower_src, lower_dst, N_NODES, buf3);
    attention(buf2, aus_l, aud_l, upper_src, upper_dst, NNZ_UP,  buf3);

    relu_kernel<<<(int)((SZ + 255) / 256), 256, 0, stream>>>(buf3, buf0, (int)SZ);
  }

  // ---- readout ----
  fill_kernel<<<1, 256, 0, stream>>>(colsum1, 0.f, CDIM);
  colsum_kernel<<<1024, 128, 0, stream>>>(buf0, colsum1, N_NODES);
  final_kernel<<<1, 32, 0, stream>>>(colsum0, colsum1, w0, b0, w1, b1, b2, (float*)d_out);
}
